// C3D_86320252715154
// MI455X (gfx1250) — compile-verified, matrix operand order checked by reading, not run on hardware
//
#include <hip/hip_runtime.h>

typedef __attribute__((ext_vector_type(16))) _Float16 v16h;
typedef __attribute__((ext_vector_type(8)))  _Float16 v8h;
typedef __attribute__((ext_vector_type(4)))  _Float16 v4h;
typedef __attribute__((ext_vector_type(8)))  float    v8f;
typedef __attribute__((ext_vector_type(4)))  float    f32x4;
typedef __attribute__((ext_vector_type(4)))  unsigned u32x4;
typedef __attribute__((ext_vector_type(8)))  int      i32x8;
typedef __attribute__((ext_vector_type(4)))  int      i32x4;

#define DEV_INLINE static __device__ __forceinline__

#if __has_builtin(__builtin_amdgcn_tensor_load_to_lds)
#define USE_TDM 1
#endif

// =====================================================================
// WMMA fragment gather from LDS (CDNA5 wave32 16-bit layout, ISA 7.12.2):
//   lane L owns row/col (L & 15); K-group base kb = 8*(L >> 4)
//   halves [0..7] <- K = kb..kb+7 ; halves [8..15] <- K = kb+16..kb+23
// Rows padded to 40 halves (80 B): 16B-aligned, conflict-light.
// =====================================================================
DEV_INLINE v16h load_frag16(const _Float16 S[][40], int row, int kb) {
  v16h f;
#pragma unroll
  for (int j = 0; j < 4; ++j) {
    f[2 * j]         = S[row][kb + 2 * j];
    f[2 * j + 1]     = S[row][kb + 2 * j + 1];
    f[8 + 2 * j]     = S[row][kb + 16 + 2 * j];
    f[8 + 2 * j + 1] = S[row][kb + 16 + 2 * j + 1];
  }
  return f;
}

template <int COUT>
DEV_INLINE void conv_epilogue(const v8f& acc, int n, int N, int THW, int HW,
                              int coBase, const float* __restrict__ bias,
                              float* __restrict__ out) {
  if (n < N) {
    int b = n / THW; int rem = n - b * THW;
    int t = rem / HW; int s = rem - t * HW;
    size_t base = (size_t)b * COUT * THW + (size_t)t * HW + s;
#pragma unroll
    for (int r = 0; r < 8; ++r) {
      float v = acc[r] + bias[coBase + r];
      out[base + (size_t)(coBase + r) * THW] = fmaxf(v, 0.f);
    }
  }
}

// =====================================================================
// Fast implicit-GEMM conv (CIN a power of two >= 32).
// K reordered as k' = tap*CIN + ci  (same permutation on A and B).
// A = pre-packed f16 weights [COUT][27*CIN] -> plain 2D tile copy: TDM.
// One K-chunk (32) lies inside a single tap -> scalar tap decode.
// Block 256 thr = 8 waves; tile 64(M) x 64(N); each wave 1x2 frags.
// =====================================================================
template <int CIN, int COUT>
__global__ __launch_bounds__(256) void conv3d_wmma_fast(
    const _Float16* __restrict__ in, const _Float16* __restrict__ wpk,
    const float* __restrict__ bias, float* __restrict__ out,
    int T, int H, int W) {
  constexpr int K = CIN * 27;
  static_assert(CIN % 32 == 0, "chunk must sit inside one tap");
  const int HW = H * W, THW = T * HW, N = 4 * THW;

  __shared__ _Float16 As[64][40];
  __shared__ _Float16 Bs[64][40];

  const int tid = threadIdx.x, lane = tid & 31, wv = tid >> 5;
  const int m0 = blockIdx.y * 64, n0 = blockIdx.x * 64;
  const int grp = lane >> 4, kb = grp * 8, r16 = lane & 15;
  const int mrow = (wv & 3) * 16 + r16;
  const int nrow0 = (wv >> 2) * 32 + r16, nrow1 = nrow0 + 16;

  // staging assignment: 4 threads per row, 8 contiguous k each
  const int srow = tid >> 2;
  const int scol = (tid & 3) * 8;

  // decode this thread's staging position ONCE (clamped, branch-free)
  const int nB  = n0 + srow;
  const bool nok = nB < N;
  const int nc  = nok ? nB : N - 1;
  const int b   = nc / THW; int rem = nc - b * THW;
  const int t   = rem / HW; int s = rem - t * HW;
  const int y   = s / W;    const int xx = s - y * W;
  const _Float16* inb = in + (size_t)b * CIN * THW;

  v8f acc0 = {}, acc1 = {};

  for (int kc = 0; kc < K; kc += 32) {
    const int tap = kc / CIN;                    // uniform -> SALU
    const int dt = tap / 9, r9 = tap - dt * 9;
    const int dy = r9 / 3,  dx = r9 - dy * 3;

    // ---- A: 64x32 f16 tile of packed weights -> LDS ----
#ifdef USE_TDM
    if (tid < 32) {  // one TDM descriptor per block per chunk (wave 0)
      unsigned long long ga =
          (unsigned long long)(uintptr_t)(wpk + (size_t)m0 * K + kc);
      unsigned lds = (unsigned)(uintptr_t)(&As[0][0]);
      u32x4 g0 = {1u,                                    // count=1 (user D#)
                  lds,                                   // lds_addr
                  (unsigned)ga,                          // global_addr[31:0]
                  (unsigned)((ga >> 32) & 0x1FFFFFFu) | (2u << 30)};  // +type=2
      i32x8 g1 = {
          (int)((1u << 16) | (1u << 20) | (3u << 22) | (3u << 25)),
          // data_size=2B | pad_enable | pad_interval=16DW | pad_amount=4DW
          (int)((unsigned)K << 16),      // tensor_dim0[15:0] (bits 63:48)
          (int)((unsigned)COUT << 16),   // dim0 hi=0 | tensor_dim1[15:0]
          (int)(32u << 16),              // tensor_dim1 hi=0 | tile_dim0=32
          64,                            // tile_dim1=64 | tile_dim2=0
          K,                             // tensor_dim0_stride[31:0]
          0, 0};
      i32x4 gz  = {0, 0, 0, 0};
      i32x8 gz8 = {0, 0, 0, 0, 0, 0, 0, 0};
      __builtin_amdgcn_tensor_load_to_lds(g0, g1, gz, gz, gz8, 0);
      __builtin_amdgcn_s_wait_tensorcnt(0);
    }
#else
    *(v8h*)&As[srow][scol] =
        *(const v8h*)(wpk + (size_t)(m0 + srow) * K + kc + scol);
#endif

    // ---- B: im2col, 8 loads stride THW, one 16B LDS store ----
    {
      const int ti = t + dt - 1, yi = y + dy - 1, xi = xx + dx - 1;
      const bool ok = nok & ((unsigned)ti < (unsigned)T) &
                      ((unsigned)yi < (unsigned)H) &
                      ((unsigned)xi < (unsigned)W);
      const int tic = ok ? ti : 0, yic = ok ? yi : 0, xic = ok ? xi : 0;
      const int ci0 = (kc & (CIN - 1)) + scol;
      const _Float16* p = inb + (size_t)ci0 * THW + (size_t)tic * HW +
                          (size_t)yic * W + xic;
      v8h hv;
#pragma unroll
      for (int j = 0; j < 8; ++j) hv[j] = p[(size_t)j * THW];
#pragma unroll
      for (int j = 0; j < 8; ++j) hv[j] = ok ? hv[j] : (_Float16)0.f;
      *(v8h*)&Bs[srow][scol] = hv;
    }
    __syncthreads();
    v16h a  = load_frag16(As, mrow, kb);
    v16h b0 = load_frag16(Bs, nrow0, kb);
    v16h b1 = load_frag16(Bs, nrow1, kb);
    acc0 = __builtin_amdgcn_wmma_f32_16x16x32_f16(false, a, false, b0,
                                                  (short)0, acc0, false, false);
    acc1 = __builtin_amdgcn_wmma_f32_16x16x32_f16(false, a, false, b1,
                                                  (short)0, acc1, false, false);
    __syncthreads();
  }

  const int coBase = m0 + (wv & 3) * 16 + grp * 8;
  conv_epilogue<COUT>(acc0, n0 + nrow0, N, THW, HW, coBase, bias, out);
  conv_epilogue<COUT>(acc1, n0 + nrow1, N, THW, HW, coBase, bias, out);
}

// =====================================================================
// Generic conv (conv1: CIN=3, K=81). Clamped branch-free staging,
// per-element k decode (tiny layer). Original K order, f32 weights.
// =====================================================================
template <int CIN, int COUT>
__global__ __launch_bounds__(256) void conv3d_wmma_gen(
    const _Float16* __restrict__ in, const float* __restrict__ wgt,
    const float* __restrict__ bias, float* __restrict__ out,
    int T, int H, int W) {
  constexpr int K = CIN * 27;
  const int HW = H * W, THW = T * HW, N = 4 * THW;

  __shared__ _Float16 As[64][40];
  __shared__ _Float16 Bs[64][40];

  const int tid = threadIdx.x, lane = tid & 31, wv = tid >> 5;
  const int m0 = blockIdx.y * 64, n0 = blockIdx.x * 64;
  const int grp = lane >> 4, kb = grp * 8, r16 = lane & 15;
  const int mrow = (wv & 3) * 16 + r16;
  const int nrow0 = (wv >> 2) * 32 + r16, nrow1 = nrow0 + 16;

  const int srow = tid >> 2;
  const int scol = (tid & 3) * 8;

  const int nB  = n0 + srow;
  const bool nok = nB < N;
  const int nc  = nok ? nB : N - 1;
  const int b   = nc / THW; int rem = nc - b * THW;
  const int t   = rem / HW; int s = rem - t * HW;
  const int y   = s / W;    const int xx = s - y * W;
  const _Float16* inb = in + (size_t)b * CIN * THW;

  v8f acc0 = {}, acc1 = {};

  for (int kc = 0; kc < K; kc += 32) {
    // ---- A: f32 weights, clamp+select ----
    {
      v8h ha;
#pragma unroll
      for (int j = 0; j < 8; ++j) {
        int k = kc + scol + j;
        int kcl = k < K ? k : K - 1;
        float v = wgt[(size_t)(m0 + srow) * K + kcl];
        ha[j] = (k < K) ? (_Float16)v : (_Float16)0.f;
      }
      *(v8h*)&As[srow][scol] = ha;
    }
    // ---- B: per-element k decode (K=81 only) ----
    {
      v8h hb;
#pragma unroll
      for (int j = 0; j < 8; ++j) {
        int k = kc + scol + j;
        int kcl = k < K ? k : K - 1;
        int ci = kcl / 27; int r2 = kcl - ci * 27;
        int dt = r2 / 9;   int r3 = r2 - dt * 9;
        int dy = r3 / 3;   int dx = r3 - dy * 3;
        int ti = t + dt - 1, yi = y + dy - 1, xi = xx + dx - 1;
        bool ok = (k < K) & nok & ((unsigned)ti < (unsigned)T) &
                  ((unsigned)yi < (unsigned)H) & ((unsigned)xi < (unsigned)W);
        int tic = ok ? ti : 0, yic = ok ? yi : 0, xic = ok ? xi : 0;
        _Float16 v = inb[(size_t)ci * THW + (size_t)tic * HW +
                         (size_t)yic * W + xic];
        hb[j] = ok ? v : (_Float16)0.f;
      }
      *(v8h*)&Bs[srow][scol] = hb;
    }
    __syncthreads();
    v16h a  = load_frag16(As, mrow, kb);
    v16h b0 = load_frag16(Bs, nrow0, kb);
    v16h b1 = load_frag16(Bs, nrow1, kb);
    acc0 = __builtin_amdgcn_wmma_f32_16x16x32_f16(false, a, false, b0,
                                                  (short)0, acc0, false, false);
    acc1 = __builtin_amdgcn_wmma_f32_16x16x32_f16(false, a, false, b1,
                                                  (short)0, acc1, false, false);
    __syncthreads();
  }

  const int coBase = m0 + (wv & 3) * 16 + grp * 8;
  conv_epilogue<COUT>(acc0, n0 + nrow0, N, THW, HW, coBase, bias, out);
  conv_epilogue<COUT>(acc1, n0 + nrow1, N, THW, HW, coBase, bias, out);
}

// =====================================================================
// FC via WMMA: out(4,N) = relu?(in(4,K) @ w(N,K)^T + b). M padded to 16.
// Branch-free vectorized staging (K % 32 == 0 for fc6/7/8).
// =====================================================================
__global__ __launch_bounds__(128) void fc_wmma(
    const float* __restrict__ in, const float* __restrict__ w,
    const float* __restrict__ bias, float* __restrict__ out,
    int K, int N, int do_relu) {
  __shared__ _Float16 As[16][40];
  __shared__ _Float16 Bs[64][40];

  const int tid = threadIdx.x, lane = tid & 31, wv = tid >> 5;
  const int n0 = blockIdx.x * 64;
  const int grp = lane >> 4, kb = grp * 8, r16 = lane & 15;

  const int arow = tid >> 3, acol = (tid & 7) * 4;   // 16 rows x 32 cols
  const int brow = tid >> 1, bcol = (tid & 1) * 16;  // 64 rows x 32 cols
  const int arcl = (arow < 4) ? arow : 0;
  const int nBr  = n0 + brow;
  const int nrcl = (nBr < N) ? nBr : N - 1;

  v8f acc = {};
  for (int kc = 0; kc < K; kc += 32) {
    {
      f32x4 f = *(const f32x4*)(in + (size_t)arcl * K + kc + acol);
      v4h h;
#pragma unroll
      for (int q = 0; q < 4; ++q)
        h[q] = (arow < 4) ? (_Float16)f[q] : (_Float16)0.f;
      *(v4h*)&As[arow][acol] = h;
    }
    {
      const float* src = w + (size_t)nrcl * K + kc + bcol;
      v8h h0, h1;
#pragma unroll
      for (int q = 0; q < 2; ++q) {
        f32x4 fa = *(const f32x4*)(src + 8 * q);
        f32x4 fb = *(const f32x4*)(src + 8 * q + 4);
        v8h* hp = q ? &h1 : &h0;
#pragma unroll
        for (int e = 0; e < 4; ++e) {
          (*hp)[e]     = (nBr < N) ? (_Float16)fa[e] : (_Float16)0.f;
          (*hp)[e + 4] = (nBr < N) ? (_Float16)fb[e] : (_Float16)0.f;
        }
      }
      *(v8h*)&Bs[brow][bcol]     = h0;
      *(v8h*)&Bs[brow][bcol + 8] = h1;
    }
    __syncthreads();
    v16h a = load_frag16(As, r16, kb);
    v16h b = load_frag16(Bs, wv * 16 + r16, kb);
    acc = __builtin_amdgcn_wmma_f32_16x16x32_f16(false, a, false, b, (short)0,
                                                 acc, false, false);
    __syncthreads();
  }
  int n = n0 + wv * 16 + r16;
  if (n < N && grp == 0) {  // batch rows M=0..3 live in grp 0, vgpr r=0..3
    float bb = bias[n];
#pragma unroll
    for (int r = 0; r < 4; ++r) {
      float v = acc[r] + bb;
      if (do_relu) v = fmaxf(v, 0.f);
      out[(size_t)r * N + n] = v;
    }
  }
}

// ============== weight pre-pack: f32 [co][ci][tap] -> f16 [co][tap*CIN+ci] ==
__global__ void pack_w_kernel(const float* __restrict__ w,
                              _Float16* __restrict__ wp, int CIN, long total) {
  int K = CIN * 27;
  long stride = (long)gridDim.x * blockDim.x;
  for (long i = (long)blockIdx.x * blockDim.x + threadIdx.x; i < total;
       i += stride) {
    long co = i / K; int k2 = (int)(i - co * K);
    int tap = k2 / CIN; int ci = k2 - tap * CIN;
    wp[i] = (_Float16)w[co * K + (size_t)ci * 27 + tap];
  }
}

// ===================== elementwise / reduction kernels =====================
__global__ void zero_kernel(float* p, int n) {
  int i = blockIdx.x * 256 + threadIdx.x;
  if (i < n) p[i] = 0.f;
}

__global__ void scale_kernel(float* p, int n, float s) {
  int i = threadIdx.x;
  if (i < n) p[i] *= s;
}

// out[t] = (|x[t]-x[t-1]| > thr) ? x[t] : x[t-1]; out[0]=x[0]   (f16 out)
__global__ void threshold_kernel(const float* __restrict__ in,
                                 _Float16* __restrict__ out,
                                 const float* __restrict__ thrRaw, int thrIdx,
                                 int T, int HW, long total) {
  float traw = thrRaw[thrIdx];
  float thr  = 20.f / (1.f + expf(-traw)) + 1.f;  // sigmoid*20+1
  long stride = (long)gridDim.x * blockDim.x;
  for (long i = (long)blockIdx.x * blockDim.x + threadIdx.x; i < total;
       i += stride) {
    int t     = (int)((i / HW) % T);
    float cur = in[i];
    float prev = (t == 0) ? cur : in[i - HW];
    float d    = cur - prev;
    out[i] = (_Float16)((fabsf(d) > thr) ? cur : prev);
  }
}

// TDVD histogram: d = (t==0) ? x[0] : x[t]-x[0]; bin = round(d)+10 in [0,21)
__global__ void hist_kernel(const float* __restrict__ x,
                            float* __restrict__ bins, int T, int HW,
                            long total) {
  __shared__ float sb[21];
  if (threadIdx.x < 21) sb[threadIdx.x] = 0.f;
  __syncthreads();
  long stride = (long)gridDim.x * blockDim.x;
  for (long i = (long)blockIdx.x * blockDim.x + threadIdx.x; i < total;
       i += stride) {
    int t   = (int)((i / HW) % T);
    float v = x[i];
    float d = (t == 0) ? v : (v - x[i - (long)t * HW]);
    float r = rintf(d);  // round-half-to-even, matches jnp.round
    if (r >= -10.f && r <= 10.f) atomicAdd(&sb[(int)r + 10], 1.0f);
  }
  __syncthreads();
  if (threadIdx.x < 21) {
    float v = sb[threadIdx.x];
    if (v != 0.f) atomicAdd(&bins[threadIdx.x], v);  // exact integer counts
  }
}

__global__ void maxpool3d_kernel(const float* __restrict__ in,
                                 float* __restrict__ out, int C, int Ti, int Hi,
                                 int Wi, int To, int Ho, int Wo, int kt, int kh,
                                 int kw, int st, int sh, int sw, int pt, int ph,
                                 int pw, long total) {
  long stride = (long)gridDim.x * blockDim.x;
  for (long i = (long)blockIdx.x * blockDim.x + threadIdx.x; i < total;
       i += stride) {
    long q  = i;
    int wo  = (int)(q % Wo); q /= Wo;
    int ho  = (int)(q % Ho); q /= Ho;
    int to  = (int)(q % To); q /= To;
    int c   = (int)(q % C);
    int b   = (int)(q / C);
    float m = -__builtin_inff();
    for (int dt = 0; dt < kt; ++dt) {
      int t = to * st - pt + dt;
      if ((unsigned)t >= (unsigned)Ti) continue;
      for (int dy = 0; dy < kh; ++dy) {
        int y = ho * sh - ph + dy;
        if ((unsigned)y >= (unsigned)Hi) continue;
        for (int dx = 0; dx < kw; ++dx) {
          int x = wo * sw - pw + dx;
          if ((unsigned)x >= (unsigned)Wi) continue;
          float v = in[(((size_t)b * C + c) * Ti + t) * Hi * Wi +
                       (size_t)y * Wi + x];
          m = fmaxf(m, v);
        }
      }
    }
    out[i] = m;
  }
}

// =====================================================================
// Host-side pipeline
// =====================================================================
static inline int gsblocks(long total) {
  long b = (total + 255) / 256;
  return (int)(b > 4096 ? 4096 : (b < 1 ? 1 : b));
}

extern "C" void kernel_launch(void* const* d_in, const int* in_sizes, int n_in,
                              void* d_out, int out_size, void* d_ws,
                              size_t ws_size, hipStream_t stream) {
  (void)in_sizes; (void)n_in; (void)ws_size;
  const float* x    = (const float*)d_in[0];
  const float* thrp = (const float*)d_in[1];
  const float* cw[8]; const float* cb[8];
  for (int i = 0; i < 8; ++i) {
    cw[i] = (const float*)d_in[2 + 2 * i];
    cb[i] = (const float*)d_in[3 + 2 * i];
  }
  const float* fc6w = (const float*)d_in[18]; const float* fc6b = (const float*)d_in[19];
  const float* fc7w = (const float*)d_in[20]; const float* fc7b = (const float*)d_in[21];
  const float* fc8w = (const float*)d_in[22]; const float* fc8b = (const float*)d_in[23];
  float* outp = (float*)d_out;

  // ws: conv-out f32 | thresh f16 | pooled f32 | packed weights f16 | fc
  char* wsb = (char*)d_ws;
  const size_t SZ_C = 205520896;  // 4*64*16*112*112*4
  const size_t SZ_T = 25690112;   // 12,845,056 elems * 2 (f16)
  const size_t SZ_P = 51380224;
  const size_t SZ_W = 55296000;   // 27,648,000 f16 packed weights
  float*    bufC = (float*)(wsb);
  _Float16* bufT = (_Float16*)(wsb + SZ_C);
  float*    bufP = (float*)(wsb + SZ_C + SZ_T);
  _Float16* bufW = (_Float16*)(wsb + SZ_C + SZ_T + SZ_P);
  float*    bufF1 = (float*)(wsb + SZ_C + SZ_T + SZ_P + SZ_W);
  float*    bufF2 = bufF1 + 4 * 4096;

  // ---- pre-pack conv2..conv5b weights to f16, k' = tap*CIN + ci ----
  const long pkoff[7] = {0, 221184, 1105920, 2875392, 6414336, 13492224, 20570112};
  const int  pcin[7]  = {64, 128, 256, 256, 512, 512, 512};
  const long pktot[7] = {221184, 884736, 1769472, 3538944, 7077888, 7077888, 7077888};
  for (int l = 0; l < 7; ++l)
    pack_w_kernel<<<gsblocks(pktot[l]), 256, 0, stream>>>(
        cw[l + 1], bufW + pkoff[l], pcin[l], pktot[l]);

  zero_kernel<<<(out_size + 255) / 256, 256, 0, stream>>>(outp, out_size);

  auto hist = [&](const float* src, int h, int C, int T, int HW) {
    long total = 4L * C * T * HW;
    float* bins = outp + 404 + h * 21;
    hist_kernel<<<gsblocks(total), 256, 0, stream>>>(src, bins, T, HW, total);
    scale_kernel<<<1, 32, 0, stream>>>(bins, 21, (float)(100.0 / (double)total));
  };
  auto thresh = [&](const float* src, int idx, int C, int T, int HW) {
    long total = 4L * C * T * HW;
    threshold_kernel<<<gsblocks(total), 256, 0, stream>>>(src, bufT, thrp, idx,
                                                          T, HW, total);
  };
  auto pool = [&](int C, int Ti, int Hi, int Wi, int To, int Ho, int Wo,
                  int kt, int kh, int kw, int st, int sh, int sw, int pt,
                  int ph, int pw) {
    long total = 4L * C * To * Ho * Wo;
    maxpool3d_kernel<<<gsblocks(total), 256, 0, stream>>>(
        bufC, bufP, C, Ti, Hi, Wi, To, Ho, Wo, kt, kh, kw, st, sh, sw, pt, ph,
        pw, total);
  };

  // ---- stage 1: conv1(3->64, 16x112x112) + pool(1,2,2) ----
  hist(x, 0, 3, 16, 112 * 112);
  thresh(x, 0, 3, 16, 112 * 112);
  conv3d_wmma_gen<3, 64><<<dim3(802816 / 64, 1), 256, 0, stream>>>(
      bufT, cw[0], cb[0], bufC, 16, 112, 112);
  pool(64, 16, 112, 112, 16, 56, 56, 1, 2, 2, 1, 2, 2, 0, 0, 0);

  // ---- stage 2: conv2(64->128, 16x56x56) + pool(2,2,2) ----
  hist(bufP, 1, 64, 16, 56 * 56);
  thresh(bufP, 1, 64, 16, 56 * 56);
  conv3d_wmma_fast<64, 128><<<dim3(200704 / 64, 2), 256, 0, stream>>>(
      bufT, bufW + pkoff[0], cb[1], bufC, 16, 56, 56);
  pool(128, 16, 56, 56, 8, 28, 28, 2, 2, 2, 2, 2, 2, 0, 0, 0);

  // ---- stage 3a ----
  hist(bufP, 2, 128, 8, 28 * 28);
  thresh(bufP, 2, 128, 8, 28 * 28);
  conv3d_wmma_fast<128, 256><<<dim3(25088 / 64, 4), 256, 0, stream>>>(
      bufT, bufW + pkoff[1], cb[2], bufC, 8, 28, 28);

  // ---- stage 3b + pool ----
  hist(bufC, 3, 256, 8, 28 * 28);
  thresh(bufC, 3, 256, 8, 28 * 28);
  conv3d_wmma_fast<256, 256><<<dim3(25088 / 64, 4), 256, 0, stream>>>(
      bufT, bufW + pkoff[2], cb[3], bufC, 8, 28, 28);
  pool(256, 8, 28, 28, 4, 14, 14, 2, 2, 2, 2, 2, 2, 0, 0, 0);

  // ---- stage 4a ----
  hist(bufP, 4, 256, 4, 14 * 14);
  thresh(bufP, 4, 256, 4, 14 * 14);
  conv3d_wmma_fast<256, 512><<<dim3(3136 / 64, 8), 256, 0, stream>>>(
      bufT, bufW + pkoff[3], cb[4], bufC, 4, 14, 14);

  // ---- stage 4b + pool ----
  hist(bufC, 5, 512, 4, 14 * 14);
  thresh(bufC, 5, 512, 4, 14 * 14);
  conv3d_wmma_fast<512, 512><<<dim3(3136 / 64, 8), 256, 0, stream>>>(
      bufT, bufW + pkoff[4], cb[5], bufC, 4, 14, 14);
  pool(512, 4, 14, 14, 2, 7, 7, 2, 2, 2, 2, 2, 2, 0, 0, 0);

  // ---- stage 5a ----
  hist(bufP, 6, 512, 2, 7 * 7);
  thresh(bufP, 6, 512, 2, 7 * 7);
  conv3d_wmma_fast<512, 512><<<dim3((392 + 63) / 64, 8), 256, 0, stream>>>(
      bufT, bufW + pkoff[5], cb[6], bufC, 2, 7, 7);

  // ---- stage 5b + pool k2 s2 pad(0,1,1) -> (512,1,4,4) ----
  hist(bufC, 7, 512, 2, 7 * 7);
  thresh(bufC, 7, 512, 2, 7 * 7);
  conv3d_wmma_fast<512, 512><<<dim3((392 + 63) / 64, 8), 256, 0, stream>>>(
      bufT, bufW + pkoff[6], cb[7], bufC, 2, 7, 7);
  pool(512, 2, 7, 7, 1, 4, 4, 2, 2, 2, 2, 2, 2, 0, 1, 1);

  // ---- FC head: bufP is (4, 8192) row-major ----
  fc_wmma<<<4096 / 64, 128, 0, stream>>>(bufP, fc6w, fc6b, bufF1, 8192, 4096, 1);
  fc_wmma<<<4096 / 64, 128, 0, stream>>>(bufF1, fc7w, fc7b, bufF2, 4096, 4096, 1);
  fc_wmma<<<(101 + 63) / 64, 128, 0, stream>>>(bufF2, fc8w, fc8b, outp, 4096, 101, 0);
}
